// MultiHeadAttentionBlock_90494960927137
// MI455X (gfx1250) — matrix operand order repaired and checked by reading, not compiled, not run on hardware
//
#include <hip/hip_runtime.h>
#include <hip/hip_bf16.h>

typedef __attribute__((ext_vector_type(16))) __bf16 v16bf;
typedef __attribute__((ext_vector_type(8)))  __bf16 v8bf;
typedef __attribute__((ext_vector_type(8)))  float  v8f;

#define EMB      768
#define HEADS    8
#define HEAD_DIM 96
#define SEQ      2048
#define BATCH    8
#define MTOT     (BATCH * SEQ)   // 16384

static __device__ inline v16bf pack16(v8bf lo, v8bf hi) {
    return __builtin_shufflevector(lo, hi, 0,1,2,3,4,5,6,7,8,9,10,11,12,13,14,15);
}
static __device__ inline v16bf load_frag(const __bf16* p0, const __bf16* p1) {
    return pack16(*(const v8bf*)p0, *(const v8bf*)p1);
}

// ---------------------------------------------------------------------------
// GEMM:  C[M,N] = A[M,768] * W[768,N] + bias
// MODE 0: A fp32 (= x), epilogue scatters bf16 into Q (scaled), K, V^T
// MODE 1: A bf16 (= attention output), epilogue writes fp32 to Out
// Block: 256 thr (8 waves), tile 128x128, K-step 32.
// Wave (wm,wn) owns 32x64 => 2x4 wmma tiles.
// ---------------------------------------------------------------------------
template<int MODE>
__global__ __launch_bounds__(256)
void gemm_wmma_kernel(const float*  __restrict__ Af32,
                      const __bf16* __restrict__ Abf,
                      const float*  __restrict__ W,
                      const float*  __restrict__ bias,
                      __bf16* __restrict__ Qd, __bf16* __restrict__ Kd,
                      __bf16* __restrict__ Vt, float* __restrict__ Out,
                      int Ndim)
{
    __shared__ __align__(16) __bf16 lA[128][40];   // [m][k], padded
    __shared__ __align__(16) __bf16 lB[128][40];   // [n][k] (transposed), padded

    const int t    = threadIdx.x;
    const int lane = t & 31;
    const int l15  = lane & 15;
    const int hi   = lane >> 4;
    const int w    = t >> 5;
    const int wm   = w & 3;        // 4 waves along M
    const int wn   = w >> 2;       // 2 waves along N
    const int m0   = blockIdx.y * 128;
    const int n0   = blockIdx.x * 128;

    v8f acc[2][4] = {};

    for (int k0 = 0; k0 < EMB; k0 += 32) {
        __syncthreads();
        if (MODE == 0) {
            // 128x32 fp32 -> bf16, 4 float4 per thread
#pragma unroll
            for (int it = 0; it < 4; ++it) {
                int idx = t + it * 256;          // 0..1023 float4 chunks
                int row = idx >> 3;
                int c   = (idx & 7) << 2;
                float4 f = *(const float4*)(Af32 + (size_t)(m0 + row) * EMB + k0 + c);
                lA[row][c + 0] = (__bf16)f.x;
                lA[row][c + 1] = (__bf16)f.y;
                lA[row][c + 2] = (__bf16)f.z;
                lA[row][c + 3] = (__bf16)f.w;
            }
        } else {
            // 128x32 bf16, 2 x 16B per thread
#pragma unroll
            for (int it = 0; it < 2; ++it) {
                int idx = t + it * 256;          // 0..511 chunks of 8 bf16
                int row = idx >> 2;
                int c   = (idx & 3) << 3;
                v8bf v = *(const v8bf*)(Abf + (size_t)(m0 + row) * EMB + k0 + c);
                *(v8bf*)&lA[row][c] = v;
            }
        }
        // W tile 32x128 fp32 -> bf16, stored transposed [n][k]
#pragma unroll
        for (int it = 0; it < 4; ++it) {
            int idx = t + it * 256;              // 0..1023 float4 chunks
            int kr  = idx >> 5;
            int c   = (idx & 31) << 2;
            float4 f = *(const float4*)(W + (size_t)(k0 + kr) * Ndim + n0 + c);
            lB[c + 0][kr] = (__bf16)f.x;
            lB[c + 1][kr] = (__bf16)f.y;
            lB[c + 2][kr] = (__bf16)f.z;
            lB[c + 3][kr] = (__bf16)f.w;
        }
        __syncthreads();

        v16bf aF[2], bF[4];
#pragma unroll
        for (int mt = 0; mt < 2; ++mt) {
            int row = wm * 32 + mt * 16 + l15;
            int kb  = hi ? 8 : 0;                // A layout: lo-half K=0-7/16-23, hi-half K=8-15/24-31
            aF[mt] = load_frag(&lA[row][kb], &lA[row][kb + 16]);
        }
#pragma unroll
        for (int nt = 0; nt < 4; ++nt) {
            int col = wn * 64 + nt * 16 + l15;
            int kb  = hi ? 8 : 0;                // B layout: lo-half K=0-7/16-23, hi-half K=8-15/24-31
            bF[nt] = load_frag(&lB[col][kb], &lB[col][kb + 16]);
        }
#pragma unroll
        for (int mt = 0; mt < 2; ++mt)
#pragma unroll
            for (int nt = 0; nt < 4; ++nt)
                acc[mt][nt] = __builtin_amdgcn_wmma_f32_16x16x32_bf16(
                    false, aF[mt], false, bF[nt], (short)0, acc[mt][nt], false, false);
    }

    // epilogue: C element (M = m0+wm*32+mt*16+r+8*hi, N = n0+wn*64+nt*16+l15) in acc[mt][nt][r]
#pragma unroll
    for (int mt = 0; mt < 2; ++mt) {
#pragma unroll
        for (int nt = 0; nt < 4; ++nt) {
            const int   n  = n0 + wn * 64 + nt * 16 + l15;
            const float bn = bias[n];
            if (MODE == 0) {
                const int which = n / EMB;
                const int rem   = n % EMB;
                const int h     = rem / HEAD_DIM;
                const int d     = rem % HEAD_DIM;
#pragma unroll
                for (int r = 0; r < 8; ++r) {
                    const int    m  = m0 + wm * 32 + mt * 16 + r + 8 * hi;
                    const int    b  = m >> 11;
                    const int    s  = m & (SEQ - 1);
                    const float  v  = acc[mt][nt][r] + bn;
                    const size_t bh = (size_t)(b * HEADS + h);
                    if (which == 0)
                        Qd[(bh * SEQ + s) * HEAD_DIM + d] = (__bf16)(v * 0.35355339059327373f); // 1/sqrt(H)
                    else if (which == 1)
                        Kd[(bh * SEQ + s) * HEAD_DIM + d] = (__bf16)v;
                    else
                        Vt[(bh * HEAD_DIM + d) * SEQ + s] = (__bf16)v;     // V stored transposed
                }
            } else {
#pragma unroll
                for (int r = 0; r < 8; ++r) {
                    const int m = m0 + wm * 32 + mt * 16 + r + 8 * hi;
                    Out[(size_t)m * EMB + n] = acc[mt][nt][r] + bn;
                }
            }
        }
    }
}

// ---------------------------------------------------------------------------
// Flash attention: grid (SEQ/64, B*H), 128 thr (4 waves). Wave = 16 Q rows.
// KV tiles of 32; Q pre-scaled by 1/sqrt(H) in the QKV kernel.
// ---------------------------------------------------------------------------
__global__ __launch_bounds__(128)
void flash_attn_kernel(const __bf16* __restrict__ Q, const __bf16* __restrict__ K,
                       const __bf16* __restrict__ Vt, __bf16* __restrict__ AO)
{
    __shared__ __align__(16) __bf16 lds_p[4][16][32];   // per-wave P tile

    const int t    = threadIdx.x;
    const int lane = t & 31;
    const int l15  = lane & 15;
    const int hi   = lane >> 4;
    const int w    = t >> 5;
    const int bh   = blockIdx.y;
    const int q0   = blockIdx.x * 64 + w * 16;

    const __bf16* Qb = Q  + (size_t)bh * SEQ * HEAD_DIM;
    const __bf16* Kb = K  + (size_t)bh * SEQ * HEAD_DIM;
    const __bf16* Vb = Vt + (size_t)bh * HEAD_DIM * SEQ;

    // Q A-fragments for the 3 K-chunks of Dh=96 (registers for whole sweep)
    v16bf aQ[3];
    {
        const int qrow = q0 + l15;
#pragma unroll
        for (int kc = 0; kc < 3; ++kc) {
            const __bf16* p = Qb + (size_t)qrow * HEAD_DIM + kc * 32 + (hi ? 8 : 0);
            aQ[kc] = load_frag(p, p + 16);
        }
    }

    float mrun[8], lrun[8];
    v8f   o[6] = {};
#pragma unroll
    for (int r = 0; r < 8; ++r) { mrun[r] = -1e30f; lrun[r] = 0.0f; }

    for (int kv0 = 0; kv0 < SEQ; kv0 += 32) {
        // ---- S = Q * K^T : two 16x16 score tiles, 3 WMMAs each ----
        v8f sc[2] = {};
#pragma unroll
        for (int nt = 0; nt < 2; ++nt) {
            const __bf16* kp = Kb + (size_t)(kv0 + nt * 16 + l15) * HEAD_DIM;
#pragma unroll
            for (int kc = 0; kc < 3; ++kc) {
                const __bf16* p = kp + kc * 32 + (hi ? 8 : 0);
                v16bf bK = load_frag(p, p + 16);
                sc[nt] = __builtin_amdgcn_wmma_f32_16x16x32_bf16(
                    false, aQ[kc], false, bK, (short)0, sc[nt], false, false);
            }
        }
        if (kv0 + 32 < SEQ) {
            __builtin_prefetch(Kb + (size_t)(kv0 + 32 + l15) * HEAD_DIM, 0, 1);
            __builtin_prefetch(Vb + (size_t)l15 * SEQ + kv0 + 32, 0, 1);
        }

        // ---- online softmax: row r+8*hi lives in sc[nt][r] across 16 lanes ----
        float p0[8], p1[8];
#pragma unroll
        for (int r = 0; r < 8; ++r) {
            float mx = fmaxf(sc[0][r], sc[1][r]);
#pragma unroll
            for (int msk = 1; msk < 16; msk <<= 1)
                mx = fmaxf(mx, __shfl_xor(mx, msk, 32));
            const float mnew  = fmaxf(mrun[r], mx);
            const float a0    = __expf(sc[0][r] - mnew);
            const float a1    = __expf(sc[1][r] - mnew);
            const float alpha = __expf(mrun[r] - mnew);
            float rs = a0 + a1;
#pragma unroll
            for (int msk = 1; msk < 16; msk <<= 1)
                rs += __shfl_xor(rs, msk, 32);
            lrun[r] = lrun[r] * alpha + rs;
            mrun[r] = mnew;
            p0[r] = a0; p1[r] = a1;
#pragma unroll
            for (int nt6 = 0; nt6 < 6; ++nt6) o[nt6][r] *= alpha;
        }

        // ---- C-layout -> A-fragment relayout of P through per-wave LDS ----
#pragma unroll
        for (int r = 0; r < 8; ++r) {
            lds_p[w][r + 8 * hi][l15]      = (__bf16)p0[r];
            lds_p[w][r + 8 * hi][16 + l15] = (__bf16)p1[r];
        }
        // same-wave DS ops are in order: reads below see the writes above
        const __bf16* pp = &lds_p[w][l15][hi ? 8 : 0];
        v16bf aP = load_frag(pp, pp + 16);

        // ---- O += P * V : 6 WMMAs (Dh = 6 x 16), V read transposed ----
#pragma unroll
        for (int nt6 = 0; nt6 < 6; ++nt6) {
            const __bf16* vp = Vb + (size_t)(nt6 * 16 + l15) * SEQ + kv0 + (hi ? 8 : 0);
            v16bf bV = load_frag(vp, vp + 16);
            o[nt6] = __builtin_amdgcn_wmma_f32_16x16x32_bf16(
                false, aP, false, bV, (short)0, o[nt6], false, false);
        }
    }

    // ---- normalize and store bf16 attention output in [B,S,E] layout ----
    const int b = bh >> 3, h = bh & 7;
#pragma unroll
    for (int r = 0; r < 8; ++r) {
        const float inv = 1.0f / lrun[r];
        const int   s   = q0 + r + 8 * hi;
#pragma unroll
        for (int nt6 = 0; nt6 < 6; ++nt6) {
            const int d = nt6 * 16 + l15;
            AO[((size_t)(b * SEQ + s)) * EMB + h * HEAD_DIM + d] = (__bf16)(o[nt6][r] * inv);
        }
    }
}

// ---------------------------------------------------------------------------
extern "C" void kernel_launch(void* const* d_in, const int* in_sizes, int n_in,
                              void* d_out, int out_size, void* d_ws, size_t ws_size,
                              hipStream_t stream) {
    const float* x      = (const float*)d_in[0];
    const float* W_qkv  = (const float*)d_in[1];
    const float* b_qkv  = (const float*)d_in[2];
    const float* W_proj = (const float*)d_in[3];
    const float* b_proj = (const float*)d_in[4];
    float* out = (float*)d_out;

    const size_t QSZ = (size_t)BATCH * HEADS * SEQ * HEAD_DIM;   // 12,582,912 elems
    __bf16* Qd = (__bf16*)d_ws;
    __bf16* Kd = Qd + QSZ;
    __bf16* Vt = Kd + QSZ;
    __bf16* AO = Vt + QSZ;      // [B,S,E] bf16; total ws use = 4*24 MB = 96 MB

    dim3 g1(3 * EMB / 128, MTOT / 128);        // (18, 128)
    gemm_wmma_kernel<0><<<g1, 256, 0, stream>>>(x, nullptr, W_qkv, b_qkv,
                                                Qd, Kd, Vt, nullptr, 3 * EMB);

    dim3 ga(SEQ / 64, BATCH * HEADS);          // (32, 64)
    flash_attn_kernel<<<ga, 128, 0, stream>>>(Qd, Kd, Vt, AO);

    dim3 g3(EMB / 128, MTOT / 128);            // (6, 128)
    gemm_wmma_kernel<1><<<g3, 256, 0, stream>>>(nullptr, AO, W_proj, b_proj,
                                                nullptr, nullptr, nullptr, out, EMB);
}
